// MAB_36206574305592
// MI455X (gfx1250) — matrix operand order repaired and checked by reading, not compiled, not run on hardware
//
#include <hip/hip_runtime.h>
#include <hip/hip_bf16.h>

typedef __attribute__((ext_vector_type(16))) _Float16 v16h;
typedef __attribute__((ext_vector_type(8)))  _Float16 v8h;
typedef __attribute__((ext_vector_type(8)))  float    v8f;

#define WMMA_F16(A, B, C) \
  __builtin_amdgcn_wmma_f32_16x16x32_f16(false, (A), false, (B), (short)0, (C), false, false)

// ---------------------------------------------------------------------------
// Fragment loaders (f16 memory -> WMMA fragments), contiguous k order, same for A and B.
// A-matrix 16x32 f16: lane L holds row M=L%16; halves j -> K = 16*hi + j
// (hi = L/16).  Two 16B LDS/global loads.
// ---------------------------------------------------------------------------
__device__ __forceinline__ v16h load_a16h(const _Float16* __restrict__ p, int hi) {
  v8h lo = *(const v8h*)(p + 16 * hi);
  v8h hh = *(const v8h*)(p + 16 * hi + 8);
  v16h a;
#pragma unroll
  for (int j = 0; j < 8; ++j) { a[j] = lo[j]; a[8 + j] = hh[j]; }
  return a;
}

// B-matrix 32x16 f16: lane L holds column N=L%16; halves j -> K = 16*(L/16)+j.
// Source must be contiguous in K per lane: p = lane's row base, 32B vector load.
__device__ __forceinline__ v16h load_b16h(const _Float16* __restrict__ p, int kg) {
  return *(const v16h*)(p + kg);
}

// ---------------------------------------------------------------------------
// Kernel 0: fp32 -> f16 conversion pre-pass (bandwidth pass, 8 elems/thread)
// ---------------------------------------------------------------------------
__global__ __launch_bounds__(256) void cvt_kernel(const float* __restrict__ src,
                                                  _Float16* __restrict__ dst, int n) {
  int i = (blockIdx.x * 256 + threadIdx.x) * 8;
  if (i < n) {
    float4 a = *(const float4*)(src + i);
    float4 b = *(const float4*)(src + i + 4);
    v8h o;
    o[0]=(_Float16)a.x; o[1]=(_Float16)a.y; o[2]=(_Float16)a.z; o[3]=(_Float16)a.w;
    o[4]=(_Float16)b.x; o[5]=(_Float16)b.y; o[6]=(_Float16)b.z; o[7]=(_Float16)b.w;
    *(v8h*)(dst + i) = o;
  }
}

// ---------------------------------------------------------------------------
// Kernel 1: projections. grid=(rows/16, DIM/256, 3), block=128 (4 waves x 16x64).
//  which=0: q = X*Wq^T+bq  -> qbuf (f32, residual) + q16 (f16 row-major)
//  which=1: k = V*Wk^T+bk  -> k16  (f16 row-major, feeds QK^T B-frags)
//  which=2: v = V*Wv^T+bv  -> vT16 (f16 [b][d][key], feeds PV B-frags)
// ---------------------------------------------------------------------------
__global__ __launch_bounds__(128) void proj_kernel(
    const _Float16* __restrict__ X16, const _Float16* __restrict__ Vi16,
    const _Float16* __restrict__ Wq16, const _Float16* __restrict__ Wk16,
    const _Float16* __restrict__ Wv16,
    const float* __restrict__ bq, const float* __restrict__ bk,
    const float* __restrict__ bv,
    float* __restrict__ qbuf, _Float16* __restrict__ q16,
    _Float16* __restrict__ k16, _Float16* __restrict__ vT16) {
  const int wave = threadIdx.x >> 5;
  const int lane = threadIdx.x & 31;
  const int m  = lane & 15;
  const int hi = lane >> 4;
  const int kg = 16 * hi;
  const int which = blockIdx.z;

  const _Float16* A    = (which == 0) ? X16 : Vi16;
  const _Float16* W    = (which == 0) ? Wq16 : ((which == 1) ? Wk16 : Wv16);
  const float*    bias = (which == 0) ? bq : ((which == 1) ? bk : bv);

  const int row0 = blockIdx.x * 16;
  const int n0   = blockIdx.y * 256 + wave * 64;

  v8f acc0 = {}, acc1 = {}, acc2 = {}, acc3 = {};
  const _Float16* arow  = A + (size_t)(row0 + m) * 512;
  const _Float16* wrow0 = W + (size_t)(n0 + 0  + m) * 512;
  const _Float16* wrow1 = W + (size_t)(n0 + 16 + m) * 512;
  const _Float16* wrow2 = W + (size_t)(n0 + 32 + m) * 512;
  const _Float16* wrow3 = W + (size_t)(n0 + 48 + m) * 512;

  for (int k0 = 0; k0 < 512; k0 += 32) {
    v16h a = load_a16h(arow + k0, hi);
    acc0 = WMMA_F16(a, load_b16h(wrow0 + k0, kg), acc0);
    acc1 = WMMA_F16(a, load_b16h(wrow1 + k0, kg), acc1);
    acc2 = WMMA_F16(a, load_b16h(wrow2 + k0, kg), acc2);
    acc3 = WMMA_F16(a, load_b16h(wrow3 + k0, kg), acc3);
  }

  v8f acc[4] = {acc0, acc1, acc2, acc3};
#pragma unroll
  for (int t = 0; t < 4; ++t) {
    const int n = n0 + t * 16 + m;
    const float bn = bias[n];
#pragma unroll
    for (int r = 0; r < 8; ++r) {
      const int row = row0 + r + 8 * hi;
      const float val = acc[t][r] + bn;
      if (which == 0) {
        qbuf[(size_t)row * 512 + n] = val;
        q16[(size_t)row * 512 + n]  = (_Float16)val;
      } else if (which == 1) {
        k16[(size_t)row * 512 + n] = (_Float16)val;
      } else {
        // transposed: [batch][dcol][key], batch = row/2048, key = row%2048
        vT16[((size_t)(row >> 11) * 512 + n) * 2048 + (row & 2047)] = (_Float16)val;
      }
    }
  }
}

// ---------------------------------------------------------------------------
// Kernel 2: flash attention. grid=(N/64, B, HEADS), block=128 (1 wave = 16 q-rows).
// Oh = qh + softmax(mask(qh kh^T / sqrt(512))) vh
// ---------------------------------------------------------------------------
__global__ __launch_bounds__(128) void attn_kernel(
    const _Float16* __restrict__ q16, const _Float16* __restrict__ k16,
    const _Float16* __restrict__ vT16, const int* __restrict__ mask,
    const float* __restrict__ qbuf, float* __restrict__ obuf,
    _Float16* __restrict__ o16) {
  __shared__ _Float16 lds_p[4][16][32];

  const int wave = threadIdx.x >> 5;
  const int lane = threadIdx.x & 31;
  const int m  = lane & 15;
  const int hi = lane >> 4;
  const int kg = 16 * hi;

  const int qt = blockIdx.x * 4 + wave;  // 0..127
  const int b  = blockIdx.y;
  const int h  = blockIdx.z;
  const float scale = 0.0441941738241592f;  // 1/sqrt(512)

  const size_t qrow0 = (size_t)b * 2048 + (size_t)qt * 16;
  const _Float16* qp = q16 + (qrow0 + m) * 512 + h * 64;
  const v16h aq0 = load_a16h(qp,      hi);  // d 0..31
  const v16h aq1 = load_a16h(qp + 32, hi);  // d 32..63

  v8f acc[4] = {};
  float row_max[8], row_sum[8];
#pragma unroll
  for (int r = 0; r < 8; ++r) { row_max[r] = -1.0e30f; row_sum[r] = 0.0f; }

  const size_t kbase = (size_t)b * 2048;
  const _Float16* vbase = vT16 + ((size_t)b * 512 + h * 64) * 2048;

  for (int c = 0; c < 2048; c += 32) {
    // ---- logits: keys [c..c+15] (s0), [c+16..c+31] (s1); B contiguous in d
    const _Float16* kr0 = k16 + (kbase + c + m) * 512 + h * 64;
    const _Float16* kr1 = kr0 + 16 * 512;
    if (c + 32 < 2048) {
      __builtin_prefetch(kr0 + 32 * 512, 0, 1);  // global_prefetch_b8
      __builtin_prefetch(kr1 + 32 * 512, 0, 1);
    }
    v8f s0 = {}, s1 = {};
    s0 = WMMA_F16(aq0, load_b16h(kr0,      kg), s0);
    s0 = WMMA_F16(aq1, load_b16h(kr0 + 32, kg), s0);
    s1 = WMMA_F16(aq0, load_b16h(kr1,      kg), s1);
    s1 = WMMA_F16(aq1, load_b16h(kr1 + 32, kg), s1);

    // ---- scale + key mask (-100 like the reference)
    const int msk0 = mask[b * 2048 + c + m];
    const int msk1 = mask[b * 2048 + c + 16 + m];
#pragma unroll
    for (int r = 0; r < 8; ++r) {
      s0[r] = msk0 ? s0[r] * scale : -100.0f;
      s1[r] = msk1 ? s1[r] * scale : -100.0f;
    }

    // ---- online softmax, row reductions across 16-lane groups
    float corr[8];
#pragma unroll
    for (int r = 0; r < 8; ++r) {
      float mx = fmaxf(s0[r], s1[r]);
      mx = fmaxf(mx, __shfl_xor(mx, 1));
      mx = fmaxf(mx, __shfl_xor(mx, 2));
      mx = fmaxf(mx, __shfl_xor(mx, 4));
      mx = fmaxf(mx, __shfl_xor(mx, 8));
      const float mnew = fmaxf(row_max[r], mx);
      corr[r] = __expf(row_max[r] - mnew);
      row_max[r] = mnew;
      const float p0 = __expf(s0[r] - mnew);
      const float p1 = __expf(s1[r] - mnew);
      float ps = p0 + p1;
      ps += __shfl_xor(ps, 1);
      ps += __shfl_xor(ps, 2);
      ps += __shfl_xor(ps, 4);
      ps += __shfl_xor(ps, 8);
      row_sum[r] = row_sum[r] * corr[r] + ps;
      lds_p[wave][r + 8 * hi][m]      = (_Float16)p0;
      lds_p[wave][r + 8 * hi][16 + m] = (_Float16)p1;
    }
#pragma unroll
    for (int t = 0; t < 4; ++t)
#pragma unroll
      for (int r = 0; r < 8; ++r) acc[t][r] *= corr[r];

    __syncthreads();  // C-layout P in LDS -> visible for A-layout reads

    const v8h plo = *(const v8h*)&lds_p[wave][m][16 * hi];
    const v8h phi = *(const v8h*)&lds_p[wave][m][16 * hi + 8];
    v16h ap;
#pragma unroll
    for (int j = 0; j < 8; ++j) { ap[j] = plo[j]; ap[8 + j] = phi[j]; }

    __syncthreads();  // reads done before next chunk overwrites LDS

    // ---- PV: B frag contiguous in key thanks to vT16 layout
#pragma unroll
    for (int t = 0; t < 4; ++t) {
      const _Float16* vrow = vbase + (size_t)(t * 16 + m) * 2048 + c;
      acc[t] = WMMA_F16(ap, load_b16h(vrow, kg), acc[t]);
    }
  }

  // ---- epilogue: normalize, residual Oh = qh + A vh, merged-head layout
#pragma unroll
  for (int r = 0; r < 8; ++r) {
    const float inv = 1.0f / row_sum[r];
    const size_t row = (qrow0 + r + 8 * hi) * 512 + h * 64 + m;
#pragma unroll
    for (int t = 0; t < 4; ++t) {
      const size_t idx = row + t * 16;
      const float o = acc[t][r] * inv + qbuf[idx];
      obuf[idx] = o;
      o16[idx]  = (_Float16)o;
    }
  }
}

// ---------------------------------------------------------------------------
// Kernel 3: out = O + relu(O*Wo^T + bo). grid=(rows/16, DIM/256), block=128.
// ---------------------------------------------------------------------------
__global__ __launch_bounds__(128) void out_kernel(
    const _Float16* __restrict__ O16, const _Float16* __restrict__ Wo16,
    const float* __restrict__ bo, const float* __restrict__ Of32,
    float* __restrict__ out) {
  const int wave = threadIdx.x >> 5;
  const int lane = threadIdx.x & 31;
  const int m  = lane & 15;
  const int hi = lane >> 4;
  const int kg = 16 * hi;

  const int row0 = blockIdx.x * 16;
  const int n0   = blockIdx.y * 256 + wave * 64;

  v8f acc0 = {}, acc1 = {}, acc2 = {}, acc3 = {};
  const _Float16* arow  = O16  + (size_t)(row0 + m) * 512;
  const _Float16* wrow0 = Wo16 + (size_t)(n0 + 0  + m) * 512;
  const _Float16* wrow1 = Wo16 + (size_t)(n0 + 16 + m) * 512;
  const _Float16* wrow2 = Wo16 + (size_t)(n0 + 32 + m) * 512;
  const _Float16* wrow3 = Wo16 + (size_t)(n0 + 48 + m) * 512;

  for (int k0 = 0; k0 < 512; k0 += 32) {
    v16h a = load_a16h(arow + k0, hi);
    acc0 = WMMA_F16(a, load_b16h(wrow0 + k0, kg), acc0);
    acc1 = WMMA_F16(a, load_b16h(wrow1 + k0, kg), acc1);
    acc2 = WMMA_F16(a, load_b16h(wrow2 + k0, kg), acc2);
    acc3 = WMMA_F16(a, load_b16h(wrow3 + k0, kg), acc3);
  }

  v8f acc[4] = {acc0, acc1, acc2, acc3};
#pragma unroll
  for (int t = 0; t < 4; ++t) {
    const int n = n0 + t * 16 + m;
    const float bn = bo[n];
#pragma unroll
    for (int r = 0; r < 8; ++r) {
      const size_t idx = (size_t)(row0 + r + 8 * hi) * 512 + n;
      out[idx] = Of32[idx] + fmaxf(0.0f, acc[t][r] + bn);
    }
  }
}

// ---------------------------------------------------------------------------
extern "C" void kernel_launch(void* const* d_in, const int* in_sizes, int n_in,
                              void* d_out, int out_size, void* d_ws, size_t ws_size,
                              hipStream_t stream) {
  const float* Q    = (const float*)d_in[0];
  const float* V    = (const float*)d_in[1];
  const int*   mask = (const int*)d_in[2];
  const float* Wq   = (const float*)d_in[3];
  const float* bq   = (const float*)d_in[4];
  const float* Wk   = (const float*)d_in[5];
  const float* bk   = (const float*)d_in[6];
  const float* Wv   = (const float*)d_in[7];
  const float* bv   = (const float*)d_in[8];
  const float* Wo   = (const float*)d_in[9];
  const float* bo   = (const float*)d_in[10];
  float* out = (float*)d_out;

  const size_t mat = (size_t)4 * 2048 * 512;   // B*N*DIM
  const size_t wsz = (size_t)512 * 512;        // DIM*DIM

  float*    qbuf = (float*)d_ws;               // f32
  float*    obuf = qbuf + mat;                 // f32
  _Float16* x16  = (_Float16*)(obuf + mat);    // f16 slabs
  _Float16* vi16 = x16 + mat;
  _Float16* q16  = vi16 + mat;
  _Float16* k16  = q16 + mat;
  _Float16* vT16 = k16 + mat;
  _Float16* o16  = vT16 + mat;
  _Float16* wq16 = o16 + mat;
  _Float16* wk16 = wq16 + wsz;
  _Float16* wv16 = wk16 + wsz;
  _Float16* wo16 = wv16 + wsz;

  // pre-pass: one-time fp32 -> f16 conversion (removes cvt from all hot loops)
  cvt_kernel<<<dim3((int)(mat / 8 / 256)), 256, 0, stream>>>(Q, x16, (int)mat);
  cvt_kernel<<<dim3((int)(mat / 8 / 256)), 256, 0, stream>>>(V, vi16, (int)mat);
  cvt_kernel<<<dim3((int)(wsz / 8 / 256)), 256, 0, stream>>>(Wq, wq16, (int)wsz);
  cvt_kernel<<<dim3((int)(wsz / 8 / 256)), 256, 0, stream>>>(Wk, wk16, (int)wsz);
  cvt_kernel<<<dim3((int)(wsz / 8 / 256)), 256, 0, stream>>>(Wv, wv16, (int)wsz);
  cvt_kernel<<<dim3((int)(wsz / 8 / 256)), 256, 0, stream>>>(Wo, wo16, (int)wsz);

  proj_kernel<<<dim3(512, 2, 3), 128, 0, stream>>>(
      x16, vi16, wq16, wk16, wv16, bq, bk, bv, qbuf, q16, k16, vT16);
  attn_kernel<<<dim3(32, 4, 8), 128, 0, stream>>>(
      q16, k16, vT16, mask, qbuf, obuf, o16);
  out_kernel<<<dim3(512, 2, 1), 128, 0, stream>>>(o16, wo16, bo, obuf, out);
}
